// S4_GAT_52604759441729
// MI455X (gfx1250) — matrix operand order repaired and checked by reading, not compiled, not run on hardware
//
#include <hip/hip_runtime.h>
#include <cstdint>
#include <cstddef>

#define B_SZ 4
#define T_LEN 354
#define TP 384            // T padded to multiple of 32
#define N_NODES 256
#define E_EDGES 8192
#define ETOT (E_EDGES + N_NODES)   // edges + self loops = 8448
#define H_DIM 128
#define NS_DIM 64
#define G_DIM 256
#define HEADS 4
#define FD 64
#define BN (B_SZ * N_NODES)        // 1024

typedef __attribute__((ext_vector_type(16))) __bf16 bf16x16;
typedef __attribute__((ext_vector_type(8))) float f32x8;
typedef __attribute__((ext_vector_type(4))) unsigned int u32x4;
typedef __attribute__((ext_vector_type(4))) int i32x4;
typedef __attribute__((ext_vector_type(8))) int i32x8;

#if __has_builtin(__builtin_amdgcn_tensor_load_to_lds) && \
    __has_builtin(__builtin_amdgcn_s_wait_tensorcnt)
#define HAVE_TDM 1
#endif

// ---------------------------------------------------------------- WMMA helpers
__device__ __forceinline__ f32x8 wmma_bf16(bf16x16 a, bf16x16 b, f32x8 c) {
  // D = A(16x32 bf16) x B(32x16 bf16) + C(16x16 f32)
  return __builtin_amdgcn_wmma_f32_16x16x32_bf16(false, a, false, b, (short)0, c,
                                                 false, false);
}

__device__ __forceinline__ f32x8 zero8() {
  f32x8 z = {0.f, 0.f, 0.f, 0.f, 0.f, 0.f, 0.f, 0.f};
  return z;
}

// A fragment (16x32, 16-bit): lane%16 = M row; lanes<16 hold K 0..15,
// lanes>=16 hold K 16..31. `row` points at (M, K=0) contiguous in K.
__device__ __forceinline__ bf16x16 load_a_bf16(const __bf16* row, int lane) {
  bf16x16 a;
  const int base = (lane < 16) ? 0 : 16;
#pragma unroll
  for (int i = 0; i < 16; ++i) a[i] = row[base + i];
  return a;
}

__device__ __forceinline__ bf16x16 load_a_f32(const float* row, int lane) {
  bf16x16 a;
  const int base = (lane < 16) ? 0 : 16;
#pragma unroll
  for (int i = 0; i < 16; ++i) a[i] = (__bf16)row[base + i];
  return a;
}

// B fragment (32x16, 16-bit): lane%16 = N col; lanes<16 hold K 0..15,
// lanes>=16 hold K 16..31. `col` points at (K=0, N) contiguous in K.
__device__ __forceinline__ bf16x16 load_b_bf16(const __bf16* col, int lane) {
  bf16x16 b;
  const int base = (lane < 16) ? 0 : 16;
#pragma unroll
  for (int i = 0; i < 16; ++i) b[i] = col[base + i];
  return b;
}

// ---------------------------------------------------------------- TDM helper
// DMA a contiguous row of 768 bf16 (1536 B) from global into LDS via the
// Tensor Data Mover (D# per CDNA5 ISA ch.8), completion via TENSORcnt.
__device__ __forceinline__ void tdm_load_row(const __bf16* gptr, __bf16* lptr) {
#ifdef HAVE_TDM
  unsigned long long ga = (unsigned long long)(const void*)gptr;
  unsigned lds = (unsigned)(unsigned long long)(void*)lptr;  // LDS byte offset
  u32x4 g0;
  g0[0] = 1u;                                   // count=1, user D#
  g0[1] = lds;                                  // lds_addr
  g0[2] = (unsigned)(ga & 0xFFFFFFFFu);         // global_addr[31:0]
  g0[3] = (unsigned)((ga >> 32) & 0x01FFFFFFu)  // global_addr[56:32]
          | (2u << 30);                         // type=2 ("image")
  i32x8 g1;
  g1[0] = (1 << 16);                            // data_size=1 -> 2 bytes/elt
  g1[1] = (int)(768u << 16);                    // tensor_dim0 = 768 (lo16)
  g1[2] = (1 << 16);                            // tensor_dim1 = 1
  g1[3] = (int)(768u << 16);                    // tile_dim0 = 768
  g1[4] = 1;                                    // tile_dim1 = 1 (tile_dim2=0)
  g1[5] = 768;                                  // tensor_dim0_stride = 768
  g1[6] = 0;
  g1[7] = 0;
  i32x4 gz = {0, 0, 0, 0};
#if defined(__clang_major__) && (__clang_major__ >= 23)
  i32x8 gz8 = {0, 0, 0, 0, 0, 0, 0, 0};
  __builtin_amdgcn_tensor_load_to_lds(g0, g1, gz, gz, gz8, 0);
#else
  __builtin_amdgcn_tensor_load_to_lds(g0, g1, gz, gz, 0);
#endif
  __builtin_amdgcn_s_wait_tensorcnt(0);
#else
  const int lane = threadIdx.x & 31;
  for (int i = lane; i < 768; i += 32) lptr[i] = gptr[i];
  __syncthreads();
#endif
}

// ---------------------------------------------------------------- scalar helpers
__device__ __forceinline__ float gelu_tanh(float x) {
  float x3 = x * x * x;
  return 0.5f * x * (1.0f + tanhf(0.7978845608f * (x + 0.044715f * x3)));
}
__device__ __forceinline__ float sigmoidf(float x) {
  return 1.0f / (1.0f + __expf(-x));
}
// monotone float<->uint key for atomicMax-based segment max
__device__ __forceinline__ unsigned fkey(float x) {
  unsigned b = __float_as_uint(x);
  return (b & 0x80000000u) ? ~b : (b | 0x80000000u);
}
__device__ __forceinline__ float fdec(unsigned k) {
  unsigned b = (k & 0x80000000u) ? (k ^ 0x80000000u) : ~k;
  return __uint_as_float(b);
}

// ---------------------------------------------------------------- prep kernels
// X[bn][t] = x[b][t][n] (bf16, zero-padded to TP)
__global__ void k_prep_x(const float* __restrict__ x, __bf16* __restrict__ Xbf) {
  int idx = blockIdx.x * blockDim.x + threadIdx.x;   // < BN*TP
  int bn = idx / TP;
  int t = idx - bn * TP;
  int b = bn >> 8, n = bn & 255;
  float v = (t < T_LEN) ? x[(b * T_LEN + t) * N_NODES + n] : 0.0f;
  Xbf[idx] = (__bf16)v;
}

// flat fp32 -> bf16 convert
__global__ void k_cvt(const float* __restrict__ in, __bf16* __restrict__ outp, int cnt) {
  int idx = blockIdx.x * blockDim.x + threadIdx.x;
  if (idx < cnt) outp[idx] = (__bf16)in[idx];
}

// W (G x G) -> Wt[j][i] = W[i][j] in bf16 (so GEMM B-fragments are K-contiguous)
__global__ void k_cvt_t(const float* __restrict__ in, __bf16* __restrict__ outp) {
  int idx = blockIdx.x * blockDim.x + threadIdx.x;   // < G*G
  int i = idx / G_DIM;
  int j = idx - i * G_DIM;
  outp[(size_t)j * G_DIM + i] = (__bf16)in[idx];
}

// ---------------------------------------------------------------- S4D kernel gen
// K[h][t] = 2*Re( sum_n Cb[h,n] * exp(dtA[h,n])^t ).
// Krev holds the REVERSED zero-padded generator row: Krev[h][383-d] = K[h][d]
// for d in [0,T), else 0, so Toeplitz B-fragments read LDS in increasing order.
// Kcum = causal prefix sum of K.
__global__ void k_s4(const float* __restrict__ ldt, const float* __restrict__ Are,
                     const float* __restrict__ Aim, const float* __restrict__ Cre,
                     const float* __restrict__ Cim, __bf16* __restrict__ Krev,
                     float* __restrict__ Kcum) {
  int h = blockIdx.x;
  int t = threadIdx.x;          // 0..TP-1
  __shared__ float sK[TP];
  __bf16* kp = Krev + (size_t)h * 768;
  kp[t] = (__bf16)0.0f;
  kp[TP + t] = (__bf16)0.0f;
  float kv = 0.0f;
  if (t < T_LEN) {
    float dt = __expf(ldt[h]);
    float acc = 0.0f;
    for (int n = 0; n < NS_DIM; ++n) {
      float ar = Are[h * NS_DIM + n], ai = Aim[h * NS_DIM + n];
      float cr = Cre[h * NS_DIM + n], ci = Cim[h * NS_DIM + n];
      float dr = ar * dt, di = ai * dt;
      float er = __expf(dr);
      float x = er * cosf(di) - 1.0f;       // exp(dtA)-1, real
      float y = er * sinf(di);              // imag
      float den = ar * ar + ai * ai;
      float br = (x * ar + y * ai) / den;   // (exp(dtA)-1)/A
      float bi = (y * ar - x * ai) / den;
      float cbr = cr * br - ci * bi;        // Cb
      float cbi = cr * bi + ci * br;
      float tf = (float)t;
      float ert = __expf(dr * tf);
      float etr = ert * cosf(di * tf);
      float eti = ert * sinf(di * tf);
      acc += cbr * etr - cbi * eti;         // Re(Cb * exp(dtA*t))
    }
    kv = 2.0f * acc;
  }
  sK[t] = (t < T_LEN) ? kv : 0.0f;
  __syncthreads();
  if (t < T_LEN) kp[(TP - 1) - t] = (__bf16)kv;   // reversed store
  if (t == 0) {
    float c = 0.0f;
    for (int i = 0; i < T_LEN; ++i) { c += sK[i]; Kcum[(size_t)h * TP + i] = c; }
    for (int i = T_LEN; i < TP; ++i) Kcum[(size_t)h * TP + i] = c;
  }
}

// ---------------------------------------------------------------- Toeplitz conv (WMMA)
// One wave owns (h, 32-row bn strip): loads the 1536B reversed K row into LDS
// once (via TDM), then loops over all 24 t-tiles, 2 WMMAs per k-step (two bn
// sub-tiles sharing the B fragment). Fused epilogue:
//   y = gelu(w*Z + b*Kcum + D*u), stored bf16 as Y[col = bn*TP+t][h].
__global__ void k_toeplitz(const __bf16* __restrict__ Xbf, const __bf16* __restrict__ Krev,
                           const float* __restrict__ Kcum, const float* __restrict__ w_in,
                           const float* __restrict__ b_in, const float* __restrict__ Dvec,
                           __bf16* __restrict__ Ybf) {
  const int lane = threadIdx.x;
  const int h = blockIdx.x;
  const int bnb = blockIdx.y * 32;
  __shared__ __bf16 sK[768];
  tdm_load_row(Krev + (size_t)h * 768, sK);

  const float wh = w_in[h], bh = b_in[h], dh = Dvec[h];
  const int kb = (lane < 16) ? 0 : 16;
  const int m16 = lane & 15;

  for (int tb = 0; tb < TP; tb += 16) {
    const int t = tb + m16;
    f32x8 acc0 = zero8();
    f32x8 acc1 = zero8();
    for (int sb = 0; sb < tb + 16; sb += 32) {   // causal: only s-tiles with s <= t
      const __bf16* arow0 = Xbf + (size_t)(bnb + m16) * TP + sb;
      const __bf16* arow1 = arow0 + (size_t)16 * TP;
      __builtin_prefetch(arow0 + 32, 0, 1);
      bf16x16 a0 = load_a_bf16(arow0, lane);
      bf16x16 a1 = load_a_bf16(arow1, lane);
      bf16x16 bb;
      const int base = (TP - 1) - t + sb + kb;   // increasing in i (reversed K)
#pragma unroll
      for (int i = 0; i < 16; ++i) bb[i] = sK[base + i];
      acc0 = wmma_bf16(a0, bb, acc0);
      acc1 = wmma_bf16(a1, bb, acc1);
    }
    const float kc = Kcum[(size_t)h * TP + ((t < T_LEN) ? t : (T_LEN - 1))];
#pragma unroll
    for (int r = 0; r < 8; ++r) {
      const int m = r + ((lane < 16) ? 0 : 8);
      {
        const int bn = bnb + m;
        float u = wh * (float)Xbf[(size_t)bn * TP + t] + bh;
        float y = gelu_tanh(wh * acc0[r] + bh * kc + dh * u);
        Ybf[(size_t)(bn * TP + t) * H_DIM + h] = (t < T_LEN) ? (__bf16)y : (__bf16)0.0f;
      }
      {
        const int bn = bnb + 16 + m;
        float u = wh * (float)Xbf[(size_t)bn * TP + t] + bh;
        float y = gelu_tanh(wh * acc1[r] + bh * kc + dh * u);
        Ybf[(size_t)(bn * TP + t) * H_DIM + h] = (t < T_LEN) ? (__bf16)y : (__bf16)0.0f;
      }
    }
  }
}

// ---------------------------------------------------------------- GLU GEMM (WMMA)
// V = Wout(2H x H) @ Y(H x BN*TP); 2 column tiles per wave, A fragments
// (value row + gate row) reused -> 4 WMMAs per k-step. Fused GLU + temporal
// mean into feats0[bn][o].
__global__ void k_glu(const __bf16* __restrict__ Woutbf, const float* __restrict__ outb,
                      const __bf16* __restrict__ Ybf, float* __restrict__ feats0) {
  const int lane = threadIdx.x;
  const int colb = blockIdx.x * 32;       // 2 adjacent 16-wide (bn,t) tiles
  const int ob = blockIdx.y * 16;         // output-channel tile, o in [0,H)
  const int bn = colb / TP;               // all 32 cols share bn (32 | TP)
  const int m16 = lane & 15;
  const int col0 = colb + m16;
  f32x8 cO[2] = {zero8(), zero8()};       // value rows, per col tile
  f32x8 cG[2] = {zero8(), zero8()};       // gate rows, per col tile
  for (int kk = 0; kk < H_DIM; kk += 32) {
    bf16x16 a1 = load_a_bf16(Woutbf + (size_t)(ob + m16) * H_DIM + kk, lane);
    bf16x16 a2 = load_a_bf16(Woutbf + (size_t)(H_DIM + ob + m16) * H_DIM + kk, lane);
    bf16x16 b0 = load_b_bf16(Ybf + (size_t)col0 * H_DIM + kk, lane);
    bf16x16 b1 = load_b_bf16(Ybf + (size_t)(col0 + 16) * H_DIM + kk, lane);
    cO[0] = wmma_bf16(a1, b0, cO[0]);
    cG[0] = wmma_bf16(a2, b0, cG[0]);
    cO[1] = wmma_bf16(a1, b1, cO[1]);
    cG[1] = wmma_bf16(a2, b1, cG[1]);
  }
#pragma unroll
  for (int cg = 0; cg < 2; ++cg) {
    const int t = colb - bn * TP + cg * 16 + m16;
    const float maskw = (t < T_LEN) ? (1.0f / (float)T_LEN) : 0.0f;
#pragma unroll
    for (int r = 0; r < 8; ++r) {
      const int o = ob + r + ((lane < 16) ? 0 : 8);
      float v1 = cO[cg][r] + outb[o];
      float v2 = cG[cg][r] + outb[H_DIM + o];
      float z = v1 * sigmoidf(v2) * maskw;
      // reduce over the 16 t-columns held by lanes 0..15 / 16..31
      z += __shfl_xor(z, 1);
      z += __shfl_xor(z, 2);
      z += __shfl_xor(z, 4);
      z += __shfl_xor(z, 8);
      if ((lane & 15) == 0) atomicAdd(&feats0[bn * H_DIM + o], z);
    }
  }
}

// ---------------------------------------------------------------- H->G projection (WMMA)
__global__ void k_proj(const float* __restrict__ feats0, const __bf16* __restrict__ Wgbf,
                       const float* __restrict__ gbias, float* __restrict__ hbuf) {
  const int lane = threadIdx.x;
  const int gb16 = blockIdx.x * 16;
  const int bnb = blockIdx.y * 16;
  f32x8 c = zero8();
  for (int kk = 0; kk < H_DIM; kk += 32) {
    bf16x16 a = load_a_f32(feats0 + (size_t)(bnb + (lane & 15)) * H_DIM + kk, lane);
    bf16x16 bb = load_b_bf16(Wgbf + (size_t)(gb16 + (lane & 15)) * H_DIM + kk, lane);
    c = wmma_bf16(a, bb, c);
  }
  const int g = gb16 + (lane & 15);
#pragma unroll
  for (int r = 0; r < 8; ++r) {
    const int bn = bnb + r + ((lane < 16) ? 0 : 8);
    hbuf[(size_t)bn * G_DIM + g] = c[r] + gbias[g];
  }
}

// ---------------------------------------------------------------- GAT GEMM (WMMA): g = h @ W
__global__ void k_gat_gemm(const float* __restrict__ hbuf, const __bf16* __restrict__ Wtbf,
                           float* __restrict__ gbuf) {
  const int lane = threadIdx.x;
  const int jb = blockIdx.x * 16;
  const int nb = blockIdx.y * 16;
  const int b = blockIdx.z;
  const float* hb = hbuf + (size_t)b * N_NODES * G_DIM;
  f32x8 c = zero8();
  for (int kk = 0; kk < G_DIM; kk += 32) {
    bf16x16 a = load_a_f32(hb + (size_t)(nb + (lane & 15)) * G_DIM + kk, lane);
    bf16x16 bb = load_b_bf16(Wtbf + (size_t)(jb + (lane & 15)) * G_DIM + kk, lane);
    c = wmma_bf16(a, bb, c);
  }
  const int j = jb + (lane & 15);
#pragma unroll
  for (int r = 0; r < 8; ++r) {
    const int n = nb + r + ((lane < 16) ? 0 : 8);
    gbuf[(size_t)(b * N_NODES + n) * G_DIM + j] = c[r];
  }
}

// ---------------------------------------------------------------- attention scalars
__global__ void k_att_scores(const float* __restrict__ gbuf, const float* __restrict__ asrc,
                             const float* __restrict__ adst, float* __restrict__ esb,
                             float* __restrict__ edb) {
  int idx = blockIdx.x * blockDim.x + threadIdx.x;   // < B*N*HEADS
  int hd = idx & (HEADS - 1);
  int bn = idx >> 2;
  const float* gp = gbuf + (size_t)bn * G_DIM + hd * FD;
  float s1 = 0.f, s2 = 0.f;
  for (int f = 0; f < FD; ++f) {
    float gv = gp[f];
    s1 += gv * asrc[hd * FD + f];
    s2 += gv * adst[hd * FD + f];
  }
  esb[idx] = s1;
  edb[idx] = s2;
}

__global__ void k_att_max(const int* __restrict__ ei, const float* __restrict__ esb,
                          const float* __restrict__ edb, unsigned* __restrict__ mkey) {
  int idx = blockIdx.x * blockDim.x + threadIdx.x;   // < B*ETOT*HEADS
  int hd = idx & (HEADS - 1);
  int rest = idx >> 2;
  int e = rest % ETOT;
  int b = rest / ETOT;
  int src = (e < E_EDGES) ? ei[e] : (e - E_EDGES);
  int dst = (e < E_EDGES) ? ei[E_EDGES + e] : (e - E_EDGES);
  float v = esb[(b * N_NODES + src) * HEADS + hd] + edb[(b * N_NODES + dst) * HEADS + hd];
  v = (v > 0.f) ? v : 0.2f * v;                      // leaky_relu(0.2)
  atomicMax(&mkey[(b * N_NODES + dst) * HEADS + hd], fkey(v));
}

__global__ void k_att_exp(const int* __restrict__ ei, const float* __restrict__ esb,
                          const float* __restrict__ edb, const unsigned* __restrict__ mkey,
                          float* __restrict__ exbuf, float* __restrict__ ssum) {
  int idx = blockIdx.x * blockDim.x + threadIdx.x;
  int hd = idx & (HEADS - 1);
  int rest = idx >> 2;
  int e = rest % ETOT;
  int b = rest / ETOT;
  int src = (e < E_EDGES) ? ei[e] : (e - E_EDGES);
  int dst = (e < E_EDGES) ? ei[E_EDGES + e] : (e - E_EDGES);
  float v = esb[(b * N_NODES + src) * HEADS + hd] + edb[(b * N_NODES + dst) * HEADS + hd];
  v = (v > 0.f) ? v : 0.2f * v;
  float m = fdec(mkey[(b * N_NODES + dst) * HEADS + hd]);
  float ex = __expf(v - m);
  exbuf[idx] = ex;
  atomicAdd(&ssum[(b * N_NODES + dst) * HEADS + hd], ex);
}

__global__ void k_att_aggr(const int* __restrict__ ei, const float* __restrict__ gbuf,
                           const float* __restrict__ exbuf, const float* __restrict__ ssum,
                           float* __restrict__ gatout) {
  int be = blockIdx.x;              // b*ETOT + e
  int b = be / ETOT;
  int e = be - b * ETOT;
  int j = threadIdx.x;
  int hd = j >> 6;
  int src = (e < E_EDGES) ? ei[e] : (e - E_EDGES);
  int dst = (e < E_EDGES) ? ei[E_EDGES + e] : (e - E_EDGES);
  float alpha = exbuf[(size_t)be * HEADS + hd] /
                (ssum[(b * N_NODES + dst) * HEADS + hd] + 1e-16f);
  atomicAdd(&gatout[(size_t)(b * N_NODES + dst) * G_DIM + j],
            gbuf[(size_t)(b * N_NODES + src) * G_DIM + j] * alpha);
}

// ---------------------------------------------------------------- GraphNorm + res + relu
__global__ void k_gnorm(const float* __restrict__ gatout, const float* __restrict__ bias,
                        const float* __restrict__ gamma, const float* __restrict__ beta,
                        const float* __restrict__ msv, float* __restrict__ hbuf) {
  const int g = blockIdx.x;
  const int b = blockIdx.y;
  const int n = threadIdx.x;        // N_NODES threads
  __shared__ float red[N_NODES];
  __shared__ float s_mu, s_var;
  const size_t idx = (size_t)(b * N_NODES + n) * G_DIM + g;
  float v = gatout[idx] + bias[g];
  float hp = hbuf[idx];
  red[n] = v;
  __syncthreads();
  for (int s = N_NODES / 2; s > 0; s >>= 1) {
    if (n < s) red[n] += red[n + s];
    __syncthreads();
  }
  if (n == 0) s_mu = red[0] * (1.0f / (float)N_NODES);
  __syncthreads();
  float c = v - msv[g] * s_mu;
  red[n] = c * c;
  __syncthreads();
  for (int s = N_NODES / 2; s > 0; s >>= 1) {
    if (n < s) red[n] += red[n + s];
    __syncthreads();
  }
  if (n == 0) s_var = red[0] * (1.0f / (float)N_NODES);
  __syncthreads();
  float hn = gamma[g] * c * rsqrtf(s_var + 1e-5f) + beta[g] + hp;
  hbuf[idx] = (hn > 0.f) ? hn : 0.f;
}

// ---------------------------------------------------------------- pool + classifier
__global__ void k_cls(const float* __restrict__ hbuf, const float* __restrict__ cw,
                      const float* __restrict__ cb, float* __restrict__ outp) {
  const int b = blockIdx.x;
  const int g = threadIdx.x;        // G_DIM threads
  float s = 0.f;
  for (int n = 0; n < N_NODES; ++n) s += hbuf[(size_t)(b * N_NODES + n) * G_DIM + g];
  s = s * (1.0f / (float)N_NODES) * cw[g];
  __shared__ float red[G_DIM];
  red[g] = s;
  __syncthreads();
  for (int st = G_DIM / 2; st > 0; st >>= 1) {
    if (g < st) red[g] += red[g + st];
    __syncthreads();
  }
  if (g == 0) outp[b] = red[0] + cb[0];
}

// ---------------------------------------------------------------- launch
extern "C" void kernel_launch(void* const* d_in, const int* in_sizes, int n_in,
                              void* d_out, int out_size, void* d_ws, size_t ws_size,
                              hipStream_t stream) {
  (void)in_sizes; (void)n_in; (void)out_size; (void)ws_size;
  const float* x     = (const float*)d_in[0];
  const int*   ei    = (const int*)d_in[1];
  const float* ipw   = (const float*)d_in[2];
  const float* ipb   = (const float*)d_in[3];
  const float* ldt   = (const float*)d_in[4];
  const float* Are   = (const float*)d_in[5];
  const float* Aim   = (const float*)d_in[6];
  const float* Cre   = (const float*)d_in[7];
  const float* Cim   = (const float*)d_in[8];
  const float* Dv    = (const float*)d_in[9];
  const float* outw  = (const float*)d_in[10];
  const float* outb  = (const float*)d_in[11];
  const float* Wg    = (const float*)d_in[12];
  const float* gbias = (const float*)d_in[13];
  const float* cw    = (const float*)d_in[35];
  const float* cb    = (const float*)d_in[36];
  float* out = (float*)d_out;

  char* wsp = (char*)d_ws;
  size_t off = 0;
  auto alloc = [&](size_t bytes) -> void* {
    void* p = wsp + off;
    off += (bytes + 255) & ~(size_t)255;
    return p;
  };
  __bf16* Xbf    = (__bf16*)alloc((size_t)BN * TP * 2);
  __bf16* Krev   = (__bf16*)alloc((size_t)H_DIM * 768 * 2);
  float*  Kcum   = (float*)alloc((size_t)H_DIM * TP * 4);
  __bf16* Woutbf = (__bf16*)alloc((size_t)2 * H_DIM * H_DIM * 2);
  __bf16* Wgbf   = (__bf16*)alloc((size_t)G_DIM * H_DIM * 2);
  __bf16* Wtbf   = (__bf16*)alloc((size_t)3 * G_DIM * G_DIM * 2);
  float*  feats0 = (float*)alloc((size_t)BN * H_DIM * 4);
  float*  hbuf   = (float*)alloc((size_t)B_SZ * N_NODES * G_DIM * 4);
  float*  gbuf   = (float*)alloc((size_t)B_SZ * N_NODES * G_DIM * 4);
  float*  esb    = (float*)alloc((size_t)B_SZ * N_NODES * HEADS * 4);
  float*  edb    = (float*)alloc((size_t)B_SZ * N_NODES * HEADS * 4);
  unsigned* mkey = (unsigned*)alloc((size_t)B_SZ * N_NODES * HEADS * 4);
  float*  ssum   = (float*)alloc((size_t)B_SZ * N_NODES * HEADS * 4);
  float*  exbuf  = (float*)alloc((size_t)B_SZ * ETOT * HEADS * 4);
  float*  gatout = (float*)alloc((size_t)B_SZ * N_NODES * G_DIM * 4);
  __bf16* Ybf    = (__bf16*)alloc((size_t)BN * TP * H_DIM * 2);

  // ---- prep
  k_prep_x<<<(BN * TP) / 256, 256, 0, stream>>>(x, Xbf);
  k_s4<<<H_DIM, TP, 0, stream>>>(ldt, Are, Aim, Cre, Cim, Krev, Kcum);
  k_cvt<<<(2 * H_DIM * H_DIM) / 256, 256, 0, stream>>>(outw, Woutbf, 2 * H_DIM * H_DIM);
  k_cvt<<<(G_DIM * H_DIM) / 256, 256, 0, stream>>>(Wg, Wgbf, G_DIM * H_DIM);
  for (int l = 0; l < 3; ++l)
    k_cvt_t<<<(G_DIM * G_DIM) / 256, 256, 0, stream>>>(
        (const float*)d_in[14 + 7 * l], Wtbf + (size_t)l * G_DIM * G_DIM);

  // ---- S4 conv (WMMA Toeplitz, TDM-fed LDS) + GLU/mean (WMMA) + projection (WMMA)
  hipMemsetAsync(feats0, 0, (size_t)BN * H_DIM * 4, stream);
  k_toeplitz<<<dim3(H_DIM, BN / 32), 32, 0, stream>>>(Xbf, Krev, Kcum, ipw, ipb, Dv, Ybf);
  k_glu<<<dim3((BN * TP) / 32, H_DIM / 16), 32, 0, stream>>>(Woutbf, outb, Ybf, feats0);
  k_proj<<<dim3(G_DIM / 16, BN / 16), 32, 0, stream>>>(feats0, Wgbf, gbias, hbuf);

  // ---- 3x GAT layers
  for (int l = 0; l < 3; ++l) {
    const float* asrc = (const float*)d_in[15 + 7 * l];
    const float* adst = (const float*)d_in[16 + 7 * l];
    const float* bias = (const float*)d_in[17 + 7 * l];
    const float* gam  = (const float*)d_in[18 + 7 * l];
    const float* bet  = (const float*)d_in[19 + 7 * l];
    const float* msv  = (const float*)d_in[20 + 7 * l];
    k_gat_gemm<<<dim3(G_DIM / 16, N_NODES / 16, B_SZ), 32, 0, stream>>>(
        hbuf, Wtbf + (size_t)l * G_DIM * G_DIM, gbuf);
    k_att_scores<<<(B_SZ * N_NODES * HEADS) / 128, 128, 0, stream>>>(gbuf, asrc, adst,
                                                                     esb, edb);
    hipMemsetAsync(mkey, 0, (size_t)B_SZ * N_NODES * HEADS * 4, stream);
    hipMemsetAsync(ssum, 0, (size_t)B_SZ * N_NODES * HEADS * 4, stream);
    hipMemsetAsync(gatout, 0, (size_t)B_SZ * N_NODES * G_DIM * 4, stream);
    k_att_max<<<(B_SZ * ETOT * HEADS) / 256, 256, 0, stream>>>(ei, esb, edb, mkey);
    k_att_exp<<<(B_SZ * ETOT * HEADS) / 256, 256, 0, stream>>>(ei, esb, edb, mkey,
                                                               exbuf, ssum);
    k_att_aggr<<<B_SZ * ETOT, 256, 0, stream>>>(ei, gbuf, exbuf, ssum, gatout);
    k_gnorm<<<dim3(G_DIM, B_SZ), N_NODES, 0, stream>>>(gatout, bias, gam, bet, msv, hbuf);
  }

  // ---- pool + classifier
  k_cls<<<B_SZ, G_DIM, 0, stream>>>(hbuf, cw, cb, out);
}
